// GAT_gc_18270790877298
// MI455X (gfx1250) — matrix operand order repaired and checked by reading, not compiled, not run on hardware
//
#include <hip/hip_runtime.h>

// ---------------- problem constants (fixed by the reference) ----------------
#define NNODES 50000
#define NEDGES 800000
#define TOTE   (NEDGES + NNODES)   // with self-loops
#define FIN    128
#define NHEAD  4
#define DHEAD  64
#define HD     256                 // NHEAD * DHEAD
#define NGRAPH 64
#define NEG_SLOPE 0.2f

typedef __attribute__((ext_vector_type(16))) _Float16 v16h;
typedef __attribute__((ext_vector_type(8)))  float    v8f;

// ---------------- helpers ----------------
__device__ __forceinline__ unsigned fkey(float f) {
    unsigned b = __float_as_uint(f);
    return (b & 0x80000000u) ? ~b : (b | 0x80000000u);
}
__device__ __forceinline__ float keyf(unsigned k) {
    return (k & 0x80000000u) ? __uint_as_float(k & 0x7FFFFFFFu) : __uint_as_float(~k);
}

__global__ void fill_f32(float* p, float v, int n) {
    int i = blockIdx.x * blockDim.x + threadIdx.x;
    if (i < n) p[i] = v;
}
__global__ void fill_u32(unsigned* p, unsigned v, int n) {
    int i = blockIdx.x * blockDim.x + threadIdx.x;
    if (i < n) p[i] = v;
}

// ---------------- pack W (fp32 [K,256]) into f16 WMMA B-fragment layout -----
// index = ((kt*16 + nt)*32 + lane)*16 + e  ->  W[kt*32 + (lane>=16?8:0) + (e&7) + (e>=8?16:0)][nt*16 + (lane&15)]
__global__ void pack_w(const float* __restrict__ W, _Float16* __restrict__ Wpk, int K) {
    int gid = blockIdx.x * blockDim.x + threadIdx.x;
    if (gid >= K * HD) return;
    int e    = gid & 15;
    int lane = (gid >> 4) & 31;
    int nt   = (gid >> 9) & 15;
    int kt   = gid >> 13;
    int k = kt * 32 + ((lane & 16) ? 8 : 0) + (e & 7) + ((e & 8) ? 16 : 0);
    int n = nt * 16 + (lane & 15);
    Wpk[gid] = (_Float16)W[k * HD + n];
}

// ---------------- GEMM: C[M=NNODES, 256] = A[M,K] * W[K,256] via WMMA -------
// block = 128 threads (4 waves). Each block: one 16-row tile, each wave: 16x64.
// K is a compile-time constant (128 or 256) so the kt loop fully unrolls and
// the scheduler can prefetch the next K-step's fragments over current WMMAs.

struct AFrag { float4 g0, g1, g2, g3; };

__device__ __forceinline__ AFrag load_a(const float* __restrict__ A, int rowK, int kb, int koff) {
    const float* Ap = A + rowK + kb + koff;
    AFrag f;
    f.g0 = *(const float4*)(Ap + 0);    // K = koff+0..3
    f.g1 = *(const float4*)(Ap + 4);    // K = koff+4..7
    f.g2 = *(const float4*)(Ap + 16);   // K = koff+16..19
    f.g3 = *(const float4*)(Ap + 20);   // K = koff+20..23
    return f;
}

__device__ __forceinline__ v16h cvt_a(const AFrag& f) {
    v16h a;
    a[0]  = (_Float16)f.g0.x; a[1]  = (_Float16)f.g0.y;
    a[2]  = (_Float16)f.g0.z; a[3]  = (_Float16)f.g0.w;
    a[4]  = (_Float16)f.g1.x; a[5]  = (_Float16)f.g1.y;
    a[6]  = (_Float16)f.g1.z; a[7]  = (_Float16)f.g1.w;
    a[8]  = (_Float16)f.g2.x; a[9]  = (_Float16)f.g2.y;
    a[10] = (_Float16)f.g2.z; a[11] = (_Float16)f.g2.w;
    a[12] = (_Float16)f.g3.x; a[13] = (_Float16)f.g3.y;
    a[14] = (_Float16)f.g3.z; a[15] = (_Float16)f.g3.w;
    return a;
}

template <int K>
__global__ __launch_bounds__(128)
void gemm_wmma(const float* __restrict__ A, const _Float16* __restrict__ Wpk,
               float* __restrict__ C) {
    const int m0   = blockIdx.x << 4;
    const int wave = threadIdx.x >> 5;       // 0..3
    const int lane = threadIdx.x & 31;
    const int n0   = wave << 6;              // wave's N start (4 tiles of 16)
    const int rowK = (m0 + (lane & 15)) * K;
    const int koff = (lane >> 4) << 3;       // 0 or 8 (A layout half-wave split)

    v8f acc0 = {}, acc1 = {}, acc2 = {}, acc3 = {};
    const v16h* Bbase = (const v16h*)Wpk + ((n0 >> 4) * 32 + lane);
    constexpr int KT = K >> 5;

    // software pipeline: prefetch fragments for step kt+1 while computing kt
    AFrag af = load_a(A, rowK, 0, koff);
    v16h b0 = Bbase[0];
    v16h b1 = Bbase[32];
    v16h b2 = Bbase[64];
    v16h b3 = Bbase[96];

#pragma unroll
    for (int kt = 0; kt < KT; ++kt) {
        AFrag af_n;
        v16h b0n, b1n, b2n, b3n;
        if (kt + 1 < KT) {
            af_n = load_a(A, rowK, (kt + 1) << 5, koff);
            const v16h* Bp = Bbase + (kt + 1) * 16 * 32;
            b0n = Bp[0];
            b1n = Bp[32];
            b2n = Bp[64];
            b3n = Bp[96];
        }
        v16h a = cvt_a(af);
        acc0 = __builtin_amdgcn_wmma_f32_16x16x32_f16(false, a, false, b0, (short)0, acc0, false, false);
        acc1 = __builtin_amdgcn_wmma_f32_16x16x32_f16(false, a, false, b1, (short)0, acc1, false, false);
        acc2 = __builtin_amdgcn_wmma_f32_16x16x32_f16(false, a, false, b2, (short)0, acc2, false, false);
        acc3 = __builtin_amdgcn_wmma_f32_16x16x32_f16(false, a, false, b3, (short)0, acc3, false, false);
        if (kt + 1 < KT) {
            af = af_n;
            b0 = b0n; b1 = b1n; b2 = b2n; b3 = b3n;
        }
    }

    // C/D layout: VGPR r -> row (r + 8*(lane>=16)), col = lane&15
    const int r0 = (lane >> 4) << 3;
    float* Cb = C + (size_t)(m0 + r0) * HD + n0 + (lane & 15);
#pragma unroll
    for (int r = 0; r < 8; ++r) {
        Cb[(size_t)r * HD +  0] = acc0[r];
        Cb[(size_t)r * HD + 16] = acc1[r];
        Cb[(size_t)r * HD + 32] = acc2[r];
        Cb[(size_t)r * HD + 48] = acc3[r];
    }
}

// ---------------- per-node attention logits --------------------------------
__global__ void attn_logits(const float* __restrict__ h,
                            const float* __restrict__ a_src,
                            const float* __restrict__ a_dst,
                            float* __restrict__ al_s, float* __restrict__ al_d) {
    int gid = blockIdx.x * blockDim.x + threadIdx.x;
    if (gid >= NNODES * NHEAD) return;
    int n  = gid >> 2;
    int hh = gid & 3;
    const float* hr = h + (size_t)n * HD + hh * DHEAD;
    const float* as = a_src + hh * DHEAD;
    const float* ad = a_dst + hh * DHEAD;
    float s1 = 0.f, s2 = 0.f;
#pragma unroll 8
    for (int d = 0; d < DHEAD; ++d) {
        float v = hr[d];
        s1 += v * as[d];
        s2 += v * ad[d];
    }
    al_s[gid] = s1;
    al_d[gid] = s2;
}

// ---------------- edge pass 1: segment max (uint-key atomicMax) ------------
__global__ void edge_max(const float* __restrict__ al_s, const float* __restrict__ al_d,
                         const long long* __restrict__ ei, unsigned* __restrict__ nmax) {
    int gid = blockIdx.x * blockDim.x + threadIdx.x;
    if (gid >= TOTE * NHEAD) return;
    int eid = gid >> 2, hh = gid & 3;
    long long s, d;
    if (eid < NEDGES) { s = ei[eid]; d = ei[NEDGES + eid]; }
    else              { s = d = eid - NEDGES; }
    float e = al_s[s * 4 + hh] + al_d[d * 4 + hh];
    e = (e >= 0.f) ? e : NEG_SLOPE * e;
    atomicMax(&nmax[d * 4 + hh], fkey(e));
}

__global__ void finalize_max(unsigned* __restrict__ nmax, int n) {
    int i = blockIdx.x * blockDim.x + threadIdx.x;
    if (i < n) ((float*)nmax)[i] = keyf(nmax[i]);
}

// ---------------- edge pass 2: softmax denominator -------------------------
__global__ void edge_denom(const float* __restrict__ al_s, const float* __restrict__ al_d,
                           const float* __restrict__ nmax, const long long* __restrict__ ei,
                           float* __restrict__ denom) {
    int gid = blockIdx.x * blockDim.x + threadIdx.x;
    if (gid >= TOTE * NHEAD) return;
    int eid = gid >> 2, hh = gid & 3;
    long long s, d;
    if (eid < NEDGES) { s = ei[eid]; d = ei[NEDGES + eid]; }
    else              { s = d = eid - NEDGES; }
    float e = al_s[s * 4 + hh] + al_d[d * 4 + hh];
    e = (e >= 0.f) ? e : NEG_SLOPE * e;
    atomicAdd(&denom[d * 4 + hh], __expf(e - nmax[d * 4 + hh]));
}

// ---------------- edge pass 3: weighted aggregation (1 wave / edge) --------
__global__ __launch_bounds__(256)
void edge_aggregate(const float* __restrict__ hG,
                    const float* __restrict__ al_s, const float* __restrict__ al_d,
                    const float* __restrict__ nmax, const float* __restrict__ denom,
                    const long long* __restrict__ ei, float* __restrict__ hO) {
    int eid = blockIdx.x * 8 + (threadIdx.x >> 5);
    if (eid >= TOTE) return;
    int lane = threadIdx.x & 31;
    long long s, d;
    if (eid < NEDGES) { s = ei[eid]; d = ei[NEDGES + eid]; }
    else              { s = d = eid - NEDGES; }
    int hh = lane >> 3;                         // 8 lanes per head (8 feats each)
    float e = al_s[s * 4 + hh] + al_d[d * 4 + hh];
    e = (e >= 0.f) ? e : NEG_SLOPE * e;
    float alpha = __expf(e - nmax[d * 4 + hh]) / (denom[d * 4 + hh] + 1e-16f);

    const float4* sp = (const float4*)(hG + (size_t)s * HD + lane * 8);
    float4 v0 = sp[0], v1 = sp[1];
    float* dp = hO + (size_t)d * HD + lane * 8;
    atomicAdd(dp + 0, v0.x * alpha);
    atomicAdd(dp + 1, v0.y * alpha);
    atomicAdd(dp + 2, v0.z * alpha);
    atomicAdd(dp + 3, v0.w * alpha);
    atomicAdd(dp + 4, v1.x * alpha);
    atomicAdd(dp + 5, v1.y * alpha);
    atomicAdd(dp + 6, v1.z * alpha);
    atomicAdd(dp + 7, v1.w * alpha);
}

// ---------------- bias + ReLU ----------------------------------------------
__global__ void bias_relu(float* __restrict__ h, const float* __restrict__ b) {
    int gid = blockIdx.x * blockDim.x + threadIdx.x;
    if (gid >= NNODES * HD) return;
    float v = h[gid] + b[gid & (HD - 1)];
    h[gid] = v > 0.f ? v : 0.f;
}

// ---------------- readout: global add pool over graph ids ------------------
__global__ void readout(const float* __restrict__ h, const long long* __restrict__ batch,
                        float* __restrict__ out) {
    int gid = blockIdx.x * blockDim.x + threadIdx.x;
    if (gid >= NNODES * HD) return;
    int n = gid >> 8, c = gid & (HD - 1);
    long long g = batch[n];
    atomicAdd(&out[g * HD + c], h[gid]);
}

// ---------------- host-side orchestration ----------------------------------
extern "C" void kernel_launch(void* const* d_in, const int* /*in_sizes*/, int /*n_in*/,
                              void* d_out, int /*out_size*/, void* d_ws, size_t /*ws_size*/,
                              hipStream_t stream) {
    const float*     x     = (const float*)d_in[0];
    const long long* ei    = (const long long*)d_in[1];
    const long long* batch = (const long long*)d_in[2];
    const float* W [3] = { (const float*)d_in[3],  (const float*)d_in[7],  (const float*)d_in[11] };
    const float* As[3] = { (const float*)d_in[4],  (const float*)d_in[8],  (const float*)d_in[12] };
    const float* Ad[3] = { (const float*)d_in[5],  (const float*)d_in[9],  (const float*)d_in[13] };
    const float* Bb[3] = { (const float*)d_in[6],  (const float*)d_in[10], (const float*)d_in[14] };
    float* out = (float*)d_out;

    // workspace layout (floats)
    float* ws = (float*)d_ws;
    const size_t NH  = (size_t)NNODES * HD;      // 12.8M
    const size_t NHh = (size_t)NNODES * NHEAD;   // 200k
    float*     hG    = ws;                        // GEMM result
    float*     hO    = ws + NH;                   // aggregated output
    float*     al_s  = ws + 2 * NH;
    float*     al_d  = al_s + NHh;
    unsigned*  nmax  = (unsigned*)(al_d + NHh);
    float*     denom = (float*)(nmax + NHh);
    _Float16*  Wpk   = (_Float16*)(denom + NHh);  // 32B-aligned (offset multiple of 32B)

    const int T = 256;
    const int gNH   = (int)((NH  + T - 1) / T);        // 50000
    const int gNHh  = (int)((NHh + T - 1) / T);
    const int gEdge = (TOTE * NHEAD + T - 1) / T;
    const int gAgg  = (TOTE + 7) / 8;

    const float* layer_in = x;
    for (int l = 0; l < 3; ++l) {
        const int K = (l == 0) ? FIN : HD;

        pack_w<<<(K * HD + T - 1) / T, T, 0, stream>>>(W[l], Wpk, K);
        if (K == FIN)
            gemm_wmma<FIN><<<NNODES / 16, 128, 0, stream>>>(layer_in, Wpk, hG);
        else
            gemm_wmma<HD><<<NNODES / 16, 128, 0, stream>>>(layer_in, Wpk, hG);
        attn_logits<<<gNHh, T, 0, stream>>>(hG, As[l], Ad[l], al_s, al_d);

        fill_u32<<<gNHh, T, 0, stream>>>(nmax, 0u, (int)NHh);
        fill_f32<<<gNHh, T, 0, stream>>>(denom, 0.f, (int)NHh);
        fill_f32<<<gNH,  T, 0, stream>>>(hO, 0.f, (int)NH);

        edge_max  <<<gEdge, T, 0, stream>>>(al_s, al_d, ei, nmax);
        finalize_max<<<gNHh, T, 0, stream>>>(nmax, (int)NHh);
        edge_denom<<<gEdge, T, 0, stream>>>(al_s, al_d, (const float*)nmax, ei, denom);
        edge_aggregate<<<gAgg, T, 0, stream>>>(hG, al_s, al_d, (const float*)nmax, denom, ei, hO);

        bias_relu<<<gNH, T, 0, stream>>>(hO, Bb[l]);
        layer_in = hO;
    }

    fill_f32<<<(NGRAPH * HD + T - 1) / T, T, 0, stream>>>(out, 0.f, NGRAPH * HD);
    readout<<<gNH, T, 0, stream>>>(hO, batch, out);
}
